// ContextAwareModel_13761075216861
// MI455X (gfx1250) — compile-verified, matrix operand order checked by reading, not run on hardware
//
#include <hip/hip_runtime.h>
#include <math.h>

// ---------------------------------------------------------------------------
// MI455X (gfx1250) implementation.
//   * Input projections x@Wih^T hoisted into big batched bf16 WMMA GEMMs.
//   * Serial T-loop keeps only h@Whh^T ([128,512]x[512,2048]) + pointwise.
//   * GEMM: 64x128 block, 8 waves, 32x32 C-tile per wave
//     (4x v_wmma_f32_16x16x32_bf16 per K-step, 2 A-frags x 2 B-frags).
//   * LDS tiles filled with GLOBAL_LOAD_ASYNC_TO_LDS_B128 (ping-pong double
//     buffer, ASYNCcnt-tracked); synchronous global->ds fallback otherwise.
// ---------------------------------------------------------------------------

typedef __attribute__((ext_vector_type(16))) __bf16 v16bf;
typedef __attribute__((ext_vector_type(8)))  __bf16 v8bf;
typedef __attribute__((ext_vector_type(8)))  float  v8f;

#define B_    128
#define T_    128
#define EMB_  1024
#define H_    512
#define H2_   1024    // 2*H
#define G4H_  2048    // 4*H
#define SRCW_ 100
#define D_    2148
#define DPAD_ 2176    // D padded to mult of 32 (K of head GEMM)
#define HALF_ 1074
#define HPAD_ 1152    // HALF padded to mult of 128 (N of head GEMM)

#if __has_builtin(__builtin_amdgcn_global_load_async_to_lds_b128)
#define HAVE_ASYNC_LDS 1
#endif

// Builtin takes pointers to GCC-vector int4 (per hipcc diagnostic), AS1/AS3.
typedef int v4i_gcc __attribute__((vector_size(16)));
typedef __attribute__((address_space(1))) v4i_gcc* gp128_t;
typedef __attribute__((address_space(3))) v4i_gcc* lp128_t;

__device__ __forceinline__ unsigned short f2bf(float f) {
  unsigned int u = __float_as_uint(f);
  unsigned int r = u + 0x7fffu + ((u >> 16) & 1u);   // round-to-nearest-even
  return (unsigned short)(r >> 16);
}
__device__ __forceinline__ float bf2f(unsigned short s) {
  return __uint_as_float(((unsigned int)s) << 16);
}
__device__ __forceinline__ float sigm(float x) { return 1.0f / (1.0f + expf(-x)); }

// Copy 16B global -> LDS. Async path uses the CDNA5 async-DMA-to-LDS op.
__device__ __forceinline__ void cp16_g2l(const __bf16* g, __bf16* l) {
#ifdef HAVE_ASYNC_LDS
  __builtin_amdgcn_global_load_async_to_lds_b128(
      (gp128_t)(void*)g, (lp128_t)(void*)l, 0, 0);
#else
  *(uint4*)l = *(const uint4*)g;
#endif
}
__device__ __forceinline__ void wait_async_all() {
#ifdef HAVE_ASYNC_LDS
#if __has_builtin(__builtin_amdgcn_s_wait_asynccnt)
  __builtin_amdgcn_s_wait_asynccnt(0);
#else
  asm volatile("s_wait_asynccnt 0x0" ::: "memory");
#endif
#endif
}

// ---------------------------------------------------------------------------
// Generic bf16 GEMM:  C[M,N] = A[M,K] * B[N,K]^T  (both row-major, K-major —
// exactly how x @ W.T presents with PyTorch-layout weights).
// Block tile 64x128, 256 threads = 8 waves, each wave -> 32x32 of C.
// Requires: M%64==0, N%128==0, K%32==0.
// ---------------------------------------------------------------------------
__global__ __launch_bounds__(256) void k_gemm_bf16_nt(
    const unsigned short* __restrict__ Au,
    const unsigned short* __restrict__ Bu,
    float* __restrict__ Cf,               // f32 out (or nullptr)
    unsigned short* __restrict__ Cb,      // bf16 out (or nullptr)
    int M, int N, int K, int ldc)
{
  constexpr int BM = 64, BN = 128, BK = 32, LDS = 40;  // +8 pad, rows 16B-aligned
  __shared__ __bf16 sA[2][BM * LDS];
  __shared__ __bf16 sB[2][BN * LDS];
  const __bf16* A  = reinterpret_cast<const __bf16*>(Au);
  const __bf16* Bm = reinterpret_cast<const __bf16*>(Bu);

  const int tid  = threadIdx.x;
  const int lane = tid & 31;
  const int wave = tid >> 5;
  const int nTiles = N / BN;
  const int m0 = (int)(blockIdx.x / nTiles) * BM;
  const int n0 = (int)(blockIdx.x % nTiles) * BN;
  const int wm = (wave & 1) * 32;         // wave's row offset in block
  const int wn = (wave >> 1) * 32;        // wave's col offset in block
  // staging chunk mapping (16B = 8 bf16 per chunk)
  const int ar = tid >> 2;                // A row 0..63
  const int ac = (tid & 3) * 8;           // A col
  const int br0 = tid >> 1;               // B row 0..127
  const int bc0 = (tid & 1) * 16;         // B cols {0..7,16..23}; +8 below
  const int r  = lane & 15;
  const int hs = (lane >> 4) * 16;        // K-half select (contiguous k = 16h + s, same for A and B)

  v8f acc00 = {}, acc01 = {}, acc10 = {}, acc11 = {};

  const int nsteps = K / BK;

  auto stage = [&](int s, int buf) {
    const int k0 = s * BK;
    cp16_g2l(A  + (size_t)(m0 + ar) * K + k0 + ac,       &sA[buf][ar * LDS + ac]);
    cp16_g2l(Bm + (size_t)(n0 + br0) * K + k0 + bc0,     &sB[buf][br0 * LDS + bc0]);
    cp16_g2l(Bm + (size_t)(n0 + br0) * K + k0 + bc0 + 8, &sB[buf][br0 * LDS + bc0 + 8]);
  };

  stage(0, 0);
  wait_async_all();
  __syncthreads();

  for (int s = 0; s < nsteps; ++s) {
    const int cur = s & 1;
    if (s + 1 < nsteps) stage(s + 1, cur ^ 1);
    if (s + 2 < nsteps) {   // speculative prefetch two tiles ahead -> L2
      const int kp = (s + 2) * BK;
      __builtin_prefetch(A  + (size_t)(m0 + ar) * K + kp + ac, 0, 1);
      __builtin_prefetch(Bm + (size_t)(n0 + br0) * K + kp + bc0, 0, 1);
    }

    union { v16bf v; v8bf h[2]; } fa0, fa1, fb0, fb1;
    fa0.h[0] = *(const v8bf*)(&sA[cur][(wm + r) * LDS + hs]);
    fa0.h[1] = *(const v8bf*)(&sA[cur][(wm + r) * LDS + hs + 8]);
    fa1.h[0] = *(const v8bf*)(&sA[cur][(wm + 16 + r) * LDS + hs]);
    fa1.h[1] = *(const v8bf*)(&sA[cur][(wm + 16 + r) * LDS + hs + 8]);
    fb0.h[0] = *(const v8bf*)(&sB[cur][(wn + r) * LDS + hs]);
    fb0.h[1] = *(const v8bf*)(&sB[cur][(wn + r) * LDS + hs + 8]);
    fb1.h[0] = *(const v8bf*)(&sB[cur][(wn + 16 + r) * LDS + hs]);
    fb1.h[1] = *(const v8bf*)(&sB[cur][(wn + 16 + r) * LDS + hs + 8]);

    acc00 = __builtin_amdgcn_wmma_f32_16x16x32_bf16(false, fa0.v, false, fb0.v,
                                                    (short)0, acc00, false, false);
    acc01 = __builtin_amdgcn_wmma_f32_16x16x32_bf16(false, fa0.v, false, fb1.v,
                                                    (short)0, acc01, false, false);
    acc10 = __builtin_amdgcn_wmma_f32_16x16x32_bf16(false, fa1.v, false, fb0.v,
                                                    (short)0, acc10, false, false);
    acc11 = __builtin_amdgcn_wmma_f32_16x16x32_bf16(false, fa1.v, false, fb1.v,
                                                    (short)0, acc11, false, false);

    wait_async_all();
    __syncthreads();
  }

  // C layout: VGPR v -> M = v + (lane>>4)*8 ; N = lane&15
  const int r0 = m0 + wm + (lane >> 4) * 8;
  const int c0 = n0 + wn + (lane & 15);
  for (int v = 0; v < 8; ++v) {
    size_t i00 = (size_t)(r0 + v) * ldc + c0;
    size_t i10 = (size_t)(r0 + 16 + v) * ldc + c0;
    if (Cf) {
      Cf[i00] = acc00[v]; Cf[i00 + 16] = acc01[v];
      Cf[i10] = acc10[v]; Cf[i10 + 16] = acc11[v];
    }
    if (Cb) {
      Cb[i00] = f2bf(acc00[v]); Cb[i00 + 16] = f2bf(acc01[v]);
      Cb[i10] = f2bf(acc10[v]); Cb[i10 + 16] = f2bf(acc11[v]);
    }
  }
}

// ---------------------------------------------------------------------------
// Elementwise helpers
// ---------------------------------------------------------------------------
__global__ __launch_bounds__(256) void k_cast_pad_bf16(
    const float* __restrict__ src, int M, int N,
    unsigned short* __restrict__ dst, int Mp, int Np)
{
  int i = blockIdx.x * blockDim.x + threadIdx.x;
  if (i >= Mp * Np) return;
  int rr = i / Np, cc = i % Np;
  float v = (rr < M && cc < N) ? src[(size_t)rr * N + cc] : 0.0f;
  dst[i] = f2bf(v);
}

__global__ __launch_bounds__(256) void k_zero_u32(unsigned int* __restrict__ p, int n)
{
  int i = blockIdx.x * blockDim.x + threadIdx.x;
  if (i < n) p[i] = 0u;
}

// Gather embedding rows for all (t,b) into bf16 X[t*B+b, EMB].
__global__ __launch_bounds__(256) void k_gather_embed(
    const int* __restrict__ article, const float* __restrict__ E,
    unsigned short* __restrict__ Xbf)
{
  int i = blockIdx.x * blockDim.x + threadIdx.x;          // one float4 each
  if (i >= T_ * B_ * (EMB_ / 4)) return;
  int tb = i / (EMB_ / 4);
  int e4 = (i % (EMB_ / 4)) * 4;
  int t = tb / B_, b = tb % B_;
  int tok = article[b * T_ + t];
  float4 v = *(const float4*)(E + (size_t)tok * EMB_ + e4);
  unsigned short* o = Xbf + (size_t)tb * EMB_ + e4;
  o[0] = f2bf(v.x); o[1] = f2bf(v.y); o[2] = f2bf(v.z); o[3] = f2bf(v.w);
}

// One LSTM direction pointwise update for one timestep.
__global__ __launch_bounds__(256) void k_lstm_pointwise(
    const float* __restrict__ gates, const unsigned short* __restrict__ Gt,
    const float* __restrict__ bih, const float* __restrict__ bhh,
    float* __restrict__ h, float* __restrict__ c,
    unsigned short* __restrict__ hbf,
    unsigned short* __restrict__ seq_out)   // nullable; row stride 2H
{
  int i = blockIdx.x * blockDim.x + threadIdx.x;
  if (i >= B_ * H_) return;
  int b = i / H_, j = i % H_;
  size_t base = (size_t)b * G4H_;
  float gi = gates[base + j]            + bf2f(Gt[base + j])            + bih[j]            + bhh[j];
  float gf = gates[base + H_ + j]       + bf2f(Gt[base + H_ + j])       + bih[H_ + j]       + bhh[H_ + j];
  float gg = gates[base + 2 * H_ + j]   + bf2f(Gt[base + 2 * H_ + j])   + bih[2 * H_ + j]   + bhh[2 * H_ + j];
  float go = gates[base + 3 * H_ + j]   + bf2f(Gt[base + 3 * H_ + j])   + bih[3 * H_ + j]   + bhh[3 * H_ + j];
  float cn = sigm(gf) * c[i] + sigm(gi) * tanhf(gg);
  float hn = sigm(go) * tanhf(cn);
  c[i] = cn; h[i] = hn;
  hbf[i] = f2bf(hn);
  if (seq_out) seq_out[(size_t)b * H2_ + j] = f2bf(hn);
}

// rep[b,:] = concat(target(1024), h2f(512), h2b(512), src_e(100), pad) bf16;
// also emits the fp32 `target` output.
__global__ __launch_bounds__(256) void k_build_rep(
    const int* __restrict__ article, const int* __restrict__ positions,
    const int* __restrict__ srcs, const float* __restrict__ E,
    const float* __restrict__ src_emb,
    const float* __restrict__ h2f, const float* __restrict__ h2b,
    unsigned short* __restrict__ repbf, float* __restrict__ out_target)
{
  int i = blockIdx.x * blockDim.x + threadIdx.x;
  if (i >= B_ * DPAD_) return;
  int b = i / DPAD_, cidx = i % DPAD_;
  float v = 0.0f;
  if (cidx < EMB_) {
    int tok = article[b * T_ + positions[b]];
    v = E[(size_t)tok * EMB_ + cidx];
    out_target[(size_t)b * EMB_ + cidx] = v;
  } else if (cidx < EMB_ + H_) {
    v = h2f[(size_t)b * H_ + (cidx - EMB_)];
  } else if (cidx < EMB_ + 2 * H_) {
    v = h2b[(size_t)b * H_ + (cidx - EMB_ - H_)];
  } else if (cidx < D_) {
    v = src_emb[(size_t)srcs[b] * SRCW_ + (cidx - EMB_ - 2 * H_)];
  }
  repbf[i] = f2bf(v);
}

__global__ __launch_bounds__(256) void k_tanh_bias(float* __restrict__ feat,
                                                   const float* __restrict__ db)
{
  int i = blockIdx.x * blockDim.x + threadIdx.x;
  if (i >= B_ * HPAD_) return;
  int n = i % HPAD_;
  feat[i] = (n < HALF_) ? tanhf(feat[i] + db[n]) : 0.0f;
}

__global__ __launch_bounds__(256) void k_head_final(
    const float* __restrict__ feat, const float* __restrict__ cW,
    const float* __restrict__ cb, float* __restrict__ logits,
    float* __restrict__ probs)
{
  int i = blockIdx.x * blockDim.x + threadIdx.x;   // B*2 = 256
  if (i >= B_ * 2) return;
  int b = i >> 1, l = i & 1;
  const float* f = feat + (size_t)b * HPAD_;
  const float* w = cW + (size_t)l * HALF_;
  float s = cb[l];
  for (int j = 0; j < HALF_; ++j) s += f[j] * w[j];
  logits[i] = s;
  probs[i] = 1.0f / (1.0f + expf(-s));
}

// ---------------------------------------------------------------------------
// Host launcher
// ---------------------------------------------------------------------------
extern "C" void kernel_launch(void* const* d_in, const int* in_sizes, int n_in,
                              void* d_out, int out_size, void* d_ws, size_t ws_size,
                              hipStream_t stream) {
  (void)in_sizes; (void)n_in; (void)out_size; (void)ws_size;

  const int*   article   = (const int*)d_in[0];
  const int*   positions = (const int*)d_in[1];
  const int*   srcs      = (const int*)d_in[2];
  const float* E         = (const float*)d_in[3];
  const float* src_emb   = (const float*)d_in[4];
  const float* Wih0f = (const float*)d_in[5];  const float* Whh0f = (const float*)d_in[6];
  const float* bih0f = (const float*)d_in[7];  const float* bhh0f = (const float*)d_in[8];
  const float* Wih0b = (const float*)d_in[9];  const float* Whh0b = (const float*)d_in[10];
  const float* bih0b = (const float*)d_in[11]; const float* bhh0b = (const float*)d_in[12];
  const float* Wih1f = (const float*)d_in[13]; const float* Whh1f = (const float*)d_in[14];
  const float* bih1f = (const float*)d_in[15]; const float* bhh1f = (const float*)d_in[16];
  const float* Wih1b = (const float*)d_in[17]; const float* Whh1b = (const float*)d_in[18];
  const float* bih1b = (const float*)d_in[19]; const float* bhh1b = (const float*)d_in[20];
  const float* dW = (const float*)d_in[21]; const float* db = (const float*)d_in[22];
  const float* cW = (const float*)d_in[23]; const float* cb = (const float*)d_in[24];

  float* out_logits = (float*)d_out;
  float* out_probs  = out_logits + B_ * 2;
  float* out_target = out_probs + B_ * 2;

  char* ws = (char*)d_ws;
  size_t off = 0;
  auto alloc = [&](size_t bytes) -> void* {
    void* p = ws + off;
    off = (off + bytes + 255) & ~(size_t)255;
    return p;
  };

  unsigned short* wih0f_bf = (unsigned short*)alloc((size_t)G4H_ * EMB_ * 2);
  unsigned short* whh0f_bf = (unsigned short*)alloc((size_t)G4H_ * H_ * 2);
  unsigned short* wih0b_bf = (unsigned short*)alloc((size_t)G4H_ * EMB_ * 2);
  unsigned short* whh0b_bf = (unsigned short*)alloc((size_t)G4H_ * H_ * 2);
  unsigned short* wih1f_bf = (unsigned short*)alloc((size_t)G4H_ * H2_ * 2);
  unsigned short* whh1f_bf = (unsigned short*)alloc((size_t)G4H_ * H_ * 2);
  unsigned short* wih1b_bf = (unsigned short*)alloc((size_t)G4H_ * H2_ * 2);
  unsigned short* whh1b_bf = (unsigned short*)alloc((size_t)G4H_ * H_ * 2);
  unsigned short* dW_bf    = (unsigned short*)alloc((size_t)HPAD_ * DPAD_ * 2);
  unsigned short* Xbf   = (unsigned short*)alloc((size_t)T_ * B_ * EMB_ * 2);
  unsigned short* Gf    = (unsigned short*)alloc((size_t)T_ * B_ * G4H_ * 2);
  unsigned short* Gb    = (unsigned short*)alloc((size_t)T_ * B_ * G4H_ * 2);
  unsigned short* h1seq = (unsigned short*)alloc((size_t)T_ * B_ * H2_ * 2);
  float* gatesF = (float*)alloc((size_t)B_ * G4H_ * 4);
  float* gatesB = (float*)alloc((size_t)B_ * G4H_ * 4);
  size_t stateStart = off;
  float* h_f = (float*)alloc((size_t)B_ * H_ * 4);
  float* c_f = (float*)alloc((size_t)B_ * H_ * 4);
  float* h_b = (float*)alloc((size_t)B_ * H_ * 4);
  float* c_b = (float*)alloc((size_t)B_ * H_ * 4);
  unsigned short* hbf_f = (unsigned short*)alloc((size_t)B_ * H_ * 2);
  unsigned short* hbf_b = (unsigned short*)alloc((size_t)B_ * H_ * 2);
  size_t stateWords = (off - stateStart) / 4;
  unsigned int* stateU32 = (unsigned int*)(ws + stateStart);
  unsigned short* repbf = (unsigned short*)alloc((size_t)B_ * DPAD_ * 2);
  float* featWS = (float*)alloc((size_t)B_ * HPAD_ * 4);

  auto cast = [&](const float* s, int M, int N, unsigned short* dst) {
    int tot = M * N;
    k_cast_pad_bf16<<<(tot + 255) / 256, 256, 0, stream>>>(s, M, N, dst, M, N);
  };
  cast(Wih0f, G4H_, EMB_, wih0f_bf);  cast(Whh0f, G4H_, H_, whh0f_bf);
  cast(Wih0b, G4H_, EMB_, wih0b_bf);  cast(Whh0b, G4H_, H_, whh0b_bf);
  cast(Wih1f, G4H_, H2_, wih1f_bf);   cast(Whh1f, G4H_, H_, whh1f_bf);
  cast(Wih1b, G4H_, H2_, wih1b_bf);   cast(Whh1b, G4H_, H_, whh1b_bf);
  k_cast_pad_bf16<<<((HPAD_ * DPAD_) + 255) / 256, 256, 0, stream>>>(
      dW, HALF_, D_, dW_bf, HPAD_, DPAD_);

  k_gather_embed<<<(T_ * B_ * (EMB_ / 4) + 255) / 256, 256, 0, stream>>>(
      article, E, Xbf);

  const int gridProj = ((T_ * B_) / 64) * (G4H_ / 128);   // 4096 blocks
  const int gridGate = (B_ / 64) * (G4H_ / 128);          // 32 blocks
  const int gridPw   = (B_ * H_) / 256;                   // 256 blocks

  auto run_layer = [&](const unsigned short* Ain,
                       const unsigned short* wihF, const unsigned short* whhF,
                       const float* biF, const float* bhF,
                       const unsigned short* wihB, const unsigned short* whhB,
                       const float* biB, const float* bhB,
                       unsigned short* seq) {
    k_zero_u32<<<((int)stateWords + 255) / 256, 256, 0, stream>>>(stateU32, (int)stateWords);
    k_gemm_bf16_nt<<<gridProj, 256, 0, stream>>>(Ain, wihF, nullptr, Gf,
                                                 T_ * B_, G4H_, EMB_, G4H_);
    k_gemm_bf16_nt<<<gridProj, 256, 0, stream>>>(Ain, wihB, nullptr, Gb,
                                                 T_ * B_, G4H_, EMB_, G4H_);
    for (int t = 0; t < T_; ++t) {
      k_gemm_bf16_nt<<<gridGate, 256, 0, stream>>>(hbf_f, whhF, gatesF, nullptr,
                                                   B_, G4H_, H_, G4H_);
      k_gemm_bf16_nt<<<gridGate, 256, 0, stream>>>(hbf_b, whhB, gatesB, nullptr,
                                                   B_, G4H_, H_, G4H_);
      k_lstm_pointwise<<<gridPw, 256, 0, stream>>>(
          gatesF, Gf + (size_t)t * B_ * G4H_, biF, bhF, h_f, c_f, hbf_f,
          seq ? seq + (size_t)t * B_ * H2_ : nullptr);
      k_lstm_pointwise<<<gridPw, 256, 0, stream>>>(
          gatesB, Gb + (size_t)t * B_ * G4H_, biB, bhB, h_b, c_b, hbf_b,
          seq ? seq + (size_t)t * B_ * H2_ + H_ : nullptr);
    }
  };

  run_layer(Xbf, wih0f_bf, whh0f_bf, bih0f, bhh0f,
            wih0b_bf, whh0b_bf, bih0b, bhh0b, h1seq);
  run_layer(h1seq, wih1f_bf, whh1f_bf, bih1f, bhh1f,
            wih1b_bf, whh1b_bf, bih1b, bhh1b, nullptr);

  k_build_rep<<<(B_ * DPAD_ + 255) / 256, 256, 0, stream>>>(
      article, positions, srcs, E, src_emb, h_f, h_b, repbf, out_target);
  k_gemm_bf16_nt<<<(B_ / 64) * (HPAD_ / 128), 256, 0, stream>>>(
      repbf, dW_bf, featWS, nullptr, B_, HPAD_, DPAD_, HPAD_);
  k_tanh_bias<<<(B_ * HPAD_ + 255) / 256, 256, 0, stream>>>(featWS, db);
  k_head_final<<<1, 256, 0, stream>>>(featWS, cW, cb, out_logits, out_probs);
}
